// FlashPhiLayer_57002805952807
// MI455X (gfx1250) — matrix operand order repaired and checked by reading, not compiled, not run on hardware
//
#include <hip/hip_runtime.h>
#include <hip/hip_bf16.h>
#include <cstdint>
#include <cstddef>

#define T_SEQ 1024
#define H_DIM 4096
#define NH_C 32
#define NKV_C 8
#define D_C 128
#define E_C 8
#define F_C 2048
#define QKV_N ((NH_C + 2 * NKV_C) * D_C) /* 6144 */
#define ATTN_SCALE 0.088388347648318447f /* 1/sqrt(128) */

#define MODE_BIAS  1
#define MODE_AUXBF 2
#define MODE_ACCUM 4

#define ATTN_SMEM_BYTES 135232

typedef __attribute__((ext_vector_type(16))) __bf16 v16bf;
typedef __attribute__((ext_vector_type(8)))  float  v8f;
typedef int v4i32 __attribute__((vector_size(16)));

union FragU { v16bf v; unsigned short us[16]; };

// ---- CDNA5 async global->LDS path (ASYNCcnt), guarded with sync fallback ----
#if defined(__has_builtin)
#if __has_builtin(__builtin_amdgcn_global_load_async_to_lds_b128)
#define USE_ASYNC_LDS 1
#endif
#endif

__device__ __forceinline__ void async_copy_b128(const void* g, void* l) {
#ifdef USE_ASYNC_LDS
  __builtin_amdgcn_global_load_async_to_lds_b128(
      (v4i32 __attribute__((address_space(1)))*)(g),
      (v4i32 __attribute__((address_space(3)))*)(l), 0, 0);
#else
  *(uint4*)l = *(const uint4*)g;
#endif
}

__device__ __forceinline__ void async_wait() {
#ifdef USE_ASYNC_LDS
  asm volatile("s_wait_asynccnt 0" ::: "memory");
#endif
}

__device__ __forceinline__ v8f v8f_zero() {
  v8f z = {0.f, 0.f, 0.f, 0.f, 0.f, 0.f, 0.f, 0.f};
  return z;
}

__device__ __forceinline__ unsigned short f2bf(float x) {
  unsigned int u = __float_as_uint(x);
  u += 0x7fffu + ((u >> 16) & 1u); // round-to-nearest-even
  return (unsigned short)(u >> 16);
}

__device__ __forceinline__ v8f wmma_bf16(v16bf a, v16bf b, v8f c) {
  // (neg_a, A, neg_b, B, c_mod, C, reuse_a, reuse_b)
  return __builtin_amdgcn_wmma_f32_16x16x32_bf16(false, a, false, b, (short)0, c,
                                                 false, false);
}

// A fragment: 16(M) x 32(K) bf16 from row-major tile[row][k], stride lda (elems).
__device__ __forceinline__ v16bf load_frag_a(const unsigned short* base, int lda,
                                             int row0, int k0) {
  int lane = (int)(threadIdx.x & 31u);
  const unsigned short* p =
      base + (size_t)(row0 + (lane & 15)) * lda + k0 + ((lane >> 4) << 4);
  FragU f;
#pragma unroll
  for (int i = 0; i < 8; ++i) f.us[i] = p[i];
#pragma unroll
  for (int i = 0; i < 8; ++i) f.us[8 + i] = p[8 + i];
  return f.v;
}

// B fragment: 32(K) x 16(N) bf16, source stored n-major: tile[n][k], stride ld.
__device__ __forceinline__ v16bf load_frag_b_nmajor(const unsigned short* base,
                                                    int ld, int n0, int k0) {
  int lane = (int)(threadIdx.x & 31u);
  const unsigned short* p =
      base + (size_t)(n0 + (lane & 15)) * ld + k0 + ((lane >> 4) << 4);
  FragU f;
#pragma unroll
  for (int i = 0; i < 16; ++i) f.us[i] = p[i];
  return f.v;
}

// B fragment: 32(K) x 16(N) bf16, source stored k-major: tile[k][n], stride ld.
__device__ __forceinline__ v16bf load_frag_b_kmajor(const unsigned short* base,
                                                    int ld, int k0, int n0) {
  int lane = (int)(threadIdx.x & 31u);
  const unsigned short* p =
      base + (size_t)(k0 + ((lane >> 4) << 4)) * ld + n0 + (lane & 15);
  FragU f;
#pragma unroll
  for (int i = 0; i < 16; ++i) f.us[i] = p[(size_t)i * ld];
  return f.v;
}

// convert 16 f32 -> 16 bf16 and store to LDS (two b128 stores)
__device__ __forceinline__ void cvt_store16(unsigned short* dst, float4 b0,
                                            float4 b1, float4 b2, float4 b3) {
  float bf_[16] = {b0.x, b0.y, b0.z, b0.w, b1.x, b1.y, b1.z, b1.w,
                   b2.x, b2.y, b2.z, b2.w, b3.x, b3.y, b3.z, b3.w};
  unsigned short tb[16];
#pragma unroll
  for (int i = 0; i < 16; ++i) tb[i] = f2bf(bf_[i]);
  *(uint4*)dst = *(uint4*)&tb[0];
  *(uint4*)(dst + 8) = *(uint4*)&tb[8];
}

// ---------------------------------------------------------------- LayerNorm
__global__ __launch_bounds__(256) void ln_kernel(const float* __restrict__ x,
                                                 const float* __restrict__ w,
                                                 const float* __restrict__ b,
                                                 unsigned short* __restrict__ out) {
  __shared__ float sa[8], sb[8];
  __shared__ float mu_s, ri_s;
  int t = blockIdx.x, tid = threadIdx.x;
  const float* xr = x + (size_t)t * H_DIM;
  float s = 0.f, s2 = 0.f;
  for (int i = tid; i < H_DIM; i += 256) {
    float v = xr[i];
    s += v;
    s2 += v * v;
  }
#pragma unroll
  for (int o = 16; o >= 1; o >>= 1) {
    s += __shfl_xor(s, o, 32);
    s2 += __shfl_xor(s2, o, 32);
  }
  if ((tid & 31) == 0) { sa[tid >> 5] = s; sb[tid >> 5] = s2; }
  __syncthreads();
  if (tid == 0) {
    float ts = 0.f, t2 = 0.f;
    for (int i = 0; i < 8; ++i) { ts += sa[i]; t2 += sb[i]; }
    float mu = ts / (float)H_DIM;
    mu_s = mu;
    ri_s = rsqrtf(t2 / (float)H_DIM - mu * mu + 1e-5f);
  }
  __syncthreads();
  float mu = mu_s, ri = ri_s;
  for (int i = tid; i < H_DIM; i += 256)
    out[(size_t)t * H_DIM + i] = f2bf((xr[i] - mu) * ri * w[i] + b[i]);
}

// ------------------------------------------------------------ Generic GEMM
// C[M,N](f32) = A_bf16[M,K] * B_f32[N,K]^T (+bias). Weights converted fp32->bf16
// on the fly. 8 waves, 64x128 tile, BK=32; double-buffered LDS; A tile staged
// with async global->LDS copies, one barrier per K step.
__global__ __launch_bounds__(256) void gemm_bf16_kernel(
    const unsigned short* __restrict__ A, int lda,
    const float* __restrict__ B, int ldb,
    float* __restrict__ C, int ldc,
    unsigned short* __restrict__ Cbf,
    const float* __restrict__ bias,
    const float* __restrict__ rowscale, int rs_stride,
    int K, int mode) {
  __shared__ unsigned short As[2][64][32];
  __shared__ unsigned short Bs[2][128][32];
  const int tid = threadIdx.x;
  const int m0 = blockIdx.y * 64;
  const int n0 = blockIdx.x * 128;
  const int wave = tid >> 5;
  const int wm = (wave >> 2) * 32;
  const int wn = (wave & 3) * 32;
  v8f acc[2][2];
#pragma unroll
  for (int i = 0; i < 2; ++i)
#pragma unroll
    for (int j = 0; j < 2; ++j) acc[i][j] = v8f_zero();

  const int ar = tid >> 2, ac = (tid & 3) << 3;   // A: 8 bf16 per thread
  const int br = tid >> 1, bc = (tid & 1) << 4;   // B: 16 f32 per thread
  const unsigned short* aptr = A + (size_t)(m0 + ar) * lda + ac;
  const float* bptr = B + (size_t)(n0 + br) * ldb + bc;

  // prologue: stage k0 = 0 into buffer 0
  async_copy_b128(aptr, &As[0][ar][ac]);
  {
    const float4* bp = (const float4*)bptr;
    cvt_store16(&Bs[0][br][bc], bp[0], bp[1], bp[2], bp[3]);
  }
  async_wait();
  __syncthreads();

  for (int k0 = 0; k0 < K; k0 += 32) {
    const int cur = (k0 >> 5) & 1;
    const int nxt = cur ^ 1;
    const bool has_next = (k0 + 32) < K;
    float4 b0, b1, b2, b3;
    if (has_next) {
      // kick next A tile straight into LDS (ASYNCcnt), fetch next B into VGPRs
      async_copy_b128(aptr + k0 + 32, &As[nxt][ar][ac]);
      const float4* bp = (const float4*)(bptr + k0 + 32);
      b0 = bp[0]; b1 = bp[1]; b2 = bp[2]; b3 = bp[3];
      __builtin_prefetch((const void*)(bptr + k0 + 64), 0, 1);
    }
    // compute current stage
#pragma unroll
    for (int mi = 0; mi < 2; ++mi) {
      v16bf af = load_frag_a(&As[cur][0][0], 32, wm + mi * 16, 0);
#pragma unroll
      for (int ni = 0; ni < 2; ++ni) {
        v16bf bfr = load_frag_b_nmajor(&Bs[cur][0][0], 32, wn + ni * 16, 0);
        acc[mi][ni] = wmma_bf16(af, bfr, acc[mi][ni]);
      }
    }
    if (has_next) cvt_store16(&Bs[nxt][br][bc], b0, b1, b2, b3);
    async_wait();
    __syncthreads();
  }

  const int lane = tid & 31;
  const int coll = lane & 15;
  const int rbase = (lane >> 4) << 3;
#pragma unroll
  for (int mi = 0; mi < 2; ++mi) {
#pragma unroll
    for (int ni = 0; ni < 2; ++ni) {
      int gm0 = m0 + wm + mi * 16 + rbase;
      int gn = n0 + wn + ni * 16 + coll;
      float bv = (mode & MODE_BIAS) ? bias[gn] : 0.f;
#pragma unroll
      for (int j = 0; j < 8; ++j) {
        int gm = gm0 + j;
        size_t off = (size_t)gm * ldc + gn;
        float v = acc[mi][ni][j];
        if (mode & MODE_ACCUM) {
          C[off] += rowscale[(size_t)gm * rs_stride] * v;
        } else {
          v += bv;
          C[off] = v;
          if (mode & MODE_AUXBF) Cbf[off] = f2bf(v);
        }
      }
    }
  }
}

// ---------------------------------------------------------------- RoPE/pack
__global__ __launch_bounds__(256) void rope_kernel(
    const float* __restrict__ qkv, const float* __restrict__ cosb,
    const float* __restrict__ sinb, unsigned short* __restrict__ qb,
    unsigned short* __restrict__ kb, unsigned short* __restrict__ vb) {
  int t = blockIdx.x, tid = threadIdx.x;
  const float* row = qkv + (size_t)t * QKV_N;
  const float* c = cosb + (size_t)t * (D_C / 2);
  const float* s = sinb + (size_t)t * (D_C / 2);
  for (int idx = tid; idx < NH_C * (D_C / 2); idx += 256) {
    int hh = idx >> 6, d = idx & 63;
    float x1 = row[hh * D_C + d], x2 = row[hh * D_C + d + 64];
    float cv = c[d], sv = s[d];
    qb[(size_t)t * (NH_C * D_C) + hh * D_C + d] = f2bf(x1 * cv - x2 * sv);
    qb[(size_t)t * (NH_C * D_C) + hh * D_C + d + 64] = f2bf(x2 * cv + x1 * sv);
  }
  const float* krow = row + NH_C * D_C;
  for (int idx = tid; idx < NKV_C * (D_C / 2); idx += 256) {
    int hh = idx >> 6, d = idx & 63;
    float x1 = krow[hh * D_C + d], x2 = krow[hh * D_C + d + 64];
    float cv = c[d], sv = s[d];
    kb[(size_t)t * (NKV_C * D_C) + hh * D_C + d] = f2bf(x1 * cv - x2 * sv);
    kb[(size_t)t * (NKV_C * D_C) + hh * D_C + d + 64] = f2bf(x2 * cv + x1 * sv);
  }
  const float* vrow = row + (NH_C + NKV_C) * D_C;
  for (int idx = tid; idx < NKV_C * D_C; idx += 256)
    vb[(size_t)t * (NKV_C * D_C) + idx] = f2bf(vrow[idx]);
}

// --------------------------------------------------------------- Attention
// Block = (head, 16 query rows), 4 waves. Full causal score row in LDS
// (CDNA5: 320 KB LDS/WGP makes the 16x1024 f32 score strip cheap).
__global__ __launch_bounds__(128) void attn_kernel(
    const unsigned short* __restrict__ Qb, const unsigned short* __restrict__ Kb,
    const unsigned short* __restrict__ Vb, unsigned short* __restrict__ Ob) {
  extern __shared__ char smem[];
  unsigned short* Qs = (unsigned short*)smem;            // 16*128 bf16
  unsigned short* Ks = Qs + 16 * 128;                    // 128*128 bf16 (4 tiles)
  float* S = (float*)(Ks + 128 * 128);                   // 16*1024 f32
  unsigned short* P = (unsigned short*)(S + 16 * 1024);  // 16*1024 bf16
  float* rowsum = (float*)(P + 16 * 1024);               // 16 f32

  const int tid = threadIdx.x;
  const int lane = tid & 31;
  const int wave = tid >> 5;
  const int qt = blockIdx.x;
  const int h = blockIdx.y;
  const int kvh = h >> 2;  // GQA 4:1
  const int q0 = qt * 16;
  const int ktmax = (q0 + 16 + 31) >> 5;  // 32-key tiles needed (causal)
  const int kspan = ktmax * 32;

  {  // stage Q tile via async copies
    int r = tid >> 3, c0 = (tid & 7) << 4;
    const unsigned short* src =
        Qb + (size_t)(q0 + r) * (NH_C * D_C) + (size_t)h * D_C + c0;
    unsigned short* dst = Qs + r * 128 + c0;
    async_copy_b128(src, dst);
    async_copy_b128(src + 8, dst + 8);
  }
  async_wait();
  __syncthreads();

  v16bf qfrag[4];
#pragma unroll
  for (int c = 0; c < 4; ++c) qfrag[c] = load_frag_a(Qs, 128, 0, c * 32);

  // ---- S = (Q K^T) * scale
  for (int kb2 = 0; kb2 < ktmax; kb2 += 4) {
    {  // stage 128 key rows (4 tiles), one row per thread, async to LDS
      const unsigned short* src =
          Kb + (size_t)(kb2 * 32 + tid) * (NKV_C * D_C) + (size_t)kvh * D_C;
      unsigned short* dst = Ks + tid * 128;
#pragma unroll
      for (int i = 0; i < 16; ++i) async_copy_b128(src + 8 * i, dst + 8 * i);
    }
    async_wait();
    __syncthreads();
    int kt = kb2 + wave;
    if (kt < ktmax) {
#pragma unroll
      for (int ni = 0; ni < 2; ++ni) {
        v8f sacc = v8f_zero();
#pragma unroll
        for (int c = 0; c < 4; ++c) {
          v16bf kf = load_frag_b_nmajor(Ks + wave * 32 * 128, 128, ni * 16, c * 32);
          sacc = wmma_bf16(qfrag[c], kf, sacc);
        }
        int coll = lane & 15, rbase = (lane >> 4) << 3;
        int scol = kt * 32 + ni * 16 + coll;
#pragma unroll
        for (int j = 0; j < 8; ++j)
          S[(rbase + j) * 1024 + scol] = sacc[j] * ATTN_SCALE;
      }
    }
    __syncthreads();
  }

  {  // causal softmax (unnormalized P; per-row sum saved)
    int row = tid >> 3, sub = tid & 7;
    int limit = q0 + row + 1;
    float mx = -1e30f;
    for (int i = sub; i < limit; i += 8) mx = fmaxf(mx, S[row * 1024 + i]);
#pragma unroll
    for (int o = 4; o >= 1; o >>= 1) mx = fmaxf(mx, __shfl_xor(mx, o, 8));
    float sum = 0.f;
    for (int i = sub; i < kspan; i += 8) {
      float p = 0.f;
      if (i < limit) {
        p = __expf(S[row * 1024 + i] - mx);
        sum += p;
      }
      P[row * 1024 + i] = f2bf(p);
    }
#pragma unroll
    for (int o = 4; o >= 1; o >>= 1) sum += __shfl_xor(sum, o, 8);
    if (sub == 0) rowsum[row] = sum;
  }
  __syncthreads();

  // ---- O = P V ; each wave owns a 32-wide d-chunk
  const int d0 = wave * 32;
  v8f oacc[2];
  oacc[0] = v8f_zero();
  oacc[1] = v8f_zero();
  for (int kb2 = 0; kb2 < ktmax; kb2 += 4) {
    {  // stage 128 value rows, async to LDS
      const unsigned short* src =
          Vb + (size_t)(kb2 * 32 + tid) * (NKV_C * D_C) + (size_t)kvh * D_C;
      unsigned short* dst = Ks + tid * 128;
#pragma unroll
      for (int i = 0; i < 16; ++i) async_copy_b128(src + 8 * i, dst + 8 * i);
    }
    async_wait();
    __syncthreads();
#pragma unroll
    for (int j = 0; j < 4; ++j) {
      int kt = kb2 + j;
      if (kt < ktmax) {
        v16bf pf = load_frag_a(P, 1024, 0, kt * 32);
        v16bf vf0 = load_frag_b_kmajor(Ks + j * 32 * 128, 128, 0, d0);
        v16bf vf1 = load_frag_b_kmajor(Ks + j * 32 * 128, 128, 0, d0 + 16);
        oacc[0] = wmma_bf16(pf, vf0, oacc[0]);
        oacc[1] = wmma_bf16(pf, vf1, oacc[1]);
      }
    }
    __syncthreads();
  }

  {  // epilogue: normalize and store bf16
    int coll = lane & 15, rbase = (lane >> 4) << 3;
#pragma unroll
    for (int ni = 0; ni < 2; ++ni) {
#pragma unroll
      for (int j = 0; j < 8; ++j) {
        int r = rbase + j;
        float v = oacc[ni][j] / rowsum[r];
        Ob[(size_t)(q0 + r) * (NH_C * D_C) + (size_t)h * D_C + d0 + ni * 16 + coll] =
            f2bf(v);
      }
    }
  }
}

// ----------------------------------------------------------------- Gating
__global__ __launch_bounds__(256) void gate_kernel(const float* __restrict__ X,
                                                   const float* __restrict__ Wg,
                                                   float* __restrict__ combine) {
  __shared__ float lg[E_C];
  int t = blockIdx.x, tid = threadIdx.x;
  int wave = tid >> 5, lane = tid & 31;
  const float* xr = X + (size_t)t * H_DIM;
  float s = 0.f;
  for (int i = lane; i < H_DIM; i += 32) s += xr[i] * Wg[(size_t)wave * H_DIM + i];
#pragma unroll
  for (int o = 16; o >= 1; o >>= 1) s += __shfl_xor(s, o, 32);
  if (lane == 0) lg[wave] = s;
  __syncthreads();
  if (tid == 0) {
    int i1 = 0;
    for (int e = 1; e < E_C; ++e)
      if (lg[e] > lg[i1]) i1 = e;
    int i2 = (i1 == 0) ? 1 : 0;
    for (int e = 0; e < E_C; ++e)
      if (e != i1 && lg[e] > lg[i2]) i2 = e;
    float r = __expf(lg[i2] - lg[i1]);  // softmax ratio, Z cancels
    float c1 = 1.f / (1.f + r), c2 = r / (1.f + r);
    for (int e = 0; e < E_C; ++e) combine[(size_t)t * E_C + e] = 0.f;
    combine[(size_t)t * E_C + i1] = c1;
    combine[(size_t)t * E_C + i2] = c2;
  }
}

// ---------------------------------------------------------------- SiLU-mul
__global__ void silu_kernel(const float* __restrict__ Hb,
                            unsigned short* __restrict__ interb) {
  int idx = blockIdx.x * blockDim.x + threadIdx.x;
  if (idx >= T_SEQ * F_C) return;
  int t = idx >> 11;  // / F_C
  int f = idx & (F_C - 1);
  float g = Hb[(size_t)t * 2 * F_C + f];
  float u = Hb[(size_t)t * 2 * F_C + F_C + f];
  float sg = g / (1.f + __expf(-g));
  interb[idx] = f2bf(sg * u);
}

// ------------------------------------------------------------------ Final
__global__ void final_kernel(const float* __restrict__ attno,
                             const float* __restrict__ moe,
                             const float* __restrict__ res,
                             float* __restrict__ out) {
  int i = blockIdx.x * blockDim.x + threadIdx.x;
  if (i >= T_SEQ * H_DIM) return;
  out[i] = attno[i] + moe[i];
  out[T_SEQ * H_DIM + i] = res[i];
}

extern "C" void kernel_launch(void* const* d_in, const int* in_sizes, int n_in,
                              void* d_out, int out_size, void* d_ws, size_t ws_size,
                              hipStream_t stream) {
  const float* hs = (const float*)d_in[0];
  const float* cosb = (const float*)d_in[1];
  const float* sinb = (const float*)d_in[2];
  const float* lnw = (const float*)d_in[3];
  const float* lnb = (const float*)d_in[4];
  const float* Wqkv = (const float*)d_in[5];
  const float* bqkv = (const float*)d_in[6];
  const float* Wo = (const float*)d_in[7];
  const float* bo = (const float*)d_in[8];
  const float* Wg = (const float*)d_in[9];
  const float* w13 = (const float*)d_in[10];
  const float* w2 = (const float*)d_in[11];
  float* out = (float*)d_out;

  char* cur = (char*)d_ws;
  auto take = [&](size_t bytes) {
    char* p = cur;
    cur += (bytes + 255) & ~(size_t)255;
    return p;
  };
  unsigned short* xn = (unsigned short*)take((size_t)T_SEQ * H_DIM * 2);
  float* qkv = (float*)take((size_t)T_SEQ * QKV_N * 4);
  unsigned short* qb = (unsigned short*)take((size_t)T_SEQ * NH_C * D_C * 2);
  unsigned short* kb = (unsigned short*)take((size_t)T_SEQ * NKV_C * D_C * 2);
  unsigned short* vb = (unsigned short*)take((size_t)T_SEQ * NKV_C * D_C * 2);
  unsigned short* attnb = (unsigned short*)take((size_t)T_SEQ * H_DIM * 2);
  float* attno = (float*)take((size_t)T_SEQ * H_DIM * 4);
  unsigned short* attnob = (unsigned short*)take((size_t)T_SEQ * H_DIM * 2);
  float* combine = (float*)take((size_t)T_SEQ * E_C * 4);
  float* h_e = (float*)take((size_t)T_SEQ * 2 * F_C * 4);
  unsigned short* interb = (unsigned short*)take((size_t)T_SEQ * F_C * 2);
  float* moe = (float*)take((size_t)T_SEQ * H_DIM * 4);

  // 1) LayerNorm -> bf16
  ln_kernel<<<T_SEQ, 256, 0, stream>>>(hs, lnw, lnb, xn);
  // 2) QKV = xn @ Wqkv^T + bqkv
  gemm_bf16_kernel<<<dim3(QKV_N / 128, T_SEQ / 64), 256, 0, stream>>>(
      xn, H_DIM, Wqkv, H_DIM, qkv, QKV_N, nullptr, bqkv, nullptr, 0, H_DIM,
      MODE_BIAS);
  // 3) RoPE + pack q/k/v to bf16
  rope_kernel<<<T_SEQ, 256, 0, stream>>>(qkv, cosb, sinb, qb, kb, vb);
  // 4) causal GQA attention
  (void)hipFuncSetAttribute(reinterpret_cast<const void*>(attn_kernel),
                            hipFuncAttributeMaxDynamicSharedMemorySize,
                            ATTN_SMEM_BYTES);
  attn_kernel<<<dim3(T_SEQ / 16, NH_C), 128, ATTN_SMEM_BYTES, stream>>>(qb, kb, vb,
                                                                        attnb);
  // 5) attn_out = attn @ Wo^T + bo  (f32 + bf16 copy)
  gemm_bf16_kernel<<<dim3(H_DIM / 128, T_SEQ / 64), 256, 0, stream>>>(
      attnb, H_DIM, Wo, H_DIM, attno, H_DIM, attnob, bo, nullptr, 0, H_DIM,
      MODE_BIAS | MODE_AUXBF);
  // 6) top-2 gating
  gate_kernel<<<T_SEQ, 256, 0, stream>>>(attno, Wg, combine);
  // 7) MoE, dense per expert (weights streamed once; matches reference math)
  (void)hipMemsetAsync(moe, 0, (size_t)T_SEQ * H_DIM * 4, stream);
  for (int e = 0; e < E_C; ++e) {
    const float* w13e = w13 + (size_t)e * 2 * F_C * H_DIM;
    const float* w2e = w2 + (size_t)e * H_DIM * F_C;
    gemm_bf16_kernel<<<dim3((2 * F_C) / 128, T_SEQ / 64), 256, 0, stream>>>(
        attnob, H_DIM, w13e, H_DIM, h_e, 2 * F_C, nullptr, nullptr, nullptr, 0,
        H_DIM, 0);
    silu_kernel<<<(T_SEQ * F_C + 255) / 256, 256, 0, stream>>>(h_e, interb);
    gemm_bf16_kernel<<<dim3(H_DIM / 128, T_SEQ / 64), 256, 0, stream>>>(
        interb, F_C, w2e, F_C, moe, H_DIM, nullptr, nullptr, combine + e, E_C, F_C,
        MODE_ACCUM);
  }
  // 8) hidden = attn_out + moe_out ; res passthrough
  final_kernel<<<(T_SEQ * H_DIM + 255) / 256, 256, 0, stream>>>(attno, moe, hs, out);
}
